// LinearAttention_12128987644626
// MI455X (gfx1250) — matrix operand order repaired and checked by reading, not compiled, not run on hardware
//
#include <hip/hip_runtime.h>

// ---------------------------------------------------------------------------
// MI455X (gfx1250) lightning-attention layer, bf16 WMMA everywhere.
// GEMM tile copies use the CDNA5 async global->LDS engine (ASYNCcnt).
// ---------------------------------------------------------------------------

typedef __bf16 bf16;
typedef __attribute__((ext_vector_type(16))) __bf16 v16bf;
typedef __attribute__((ext_vector_type(8)))  __bf16 v8bf;
typedef __attribute__((ext_vector_type(8)))  float  v8f;

#define EMBED   2048
#define HEADS   16
#define HD      128
#define SEQ     2048
#define NBATCH  2
#define ROWS    (NBATCH*SEQ)      // 4096
#define CHUNKC  128
#define NCHUNK  (SEQ/CHUNKC)      // 16

#define WMMA_BF16(a,b,c) \
  __builtin_amdgcn_wmma_f32_16x16x32_bf16(false,(a),false,(b),(short)0,(c),false,false)

// A-matrix fragment (16x32, M rows): lane<16: K=0..7,16..23 ; lane>=16: K=8..15,24..31
__device__ __forceinline__ v16bf frag_ldA(const bf16* p, int hi) {
  v8bf lo = *(const v8bf*)(p + hi * 8);
  v8bf hh = *(const v8bf*)(p + 16 + hi * 8);
  return __builtin_shufflevector(lo, hh, 0,1,2,3,4,5,6,7,8,9,10,11,12,13,14,15);
}
// B-matrix fragment (32x16, N cols): lane<16: K=0..7,16..23 ; lane>=16: K=8..15,24..31
__device__ __forceinline__ v16bf frag_ldB(const bf16* p, int hi) {
  v8bf lo = *(const v8bf*)(p + hi * 8);
  v8bf hh = *(const v8bf*)(p + 16 + hi * 8);
  return __builtin_shufflevector(lo, hh, 0,1,2,3,4,5,6,7,8,9,10,11,12,13,14,15);
}

// CDNA5 async global->LDS copy, 16B per lane. VDST = LDS byte address,
// VADDR = 64-bit global address (GV mode). Tracked by ASYNCcnt.
__device__ __forceinline__ void async_copy_b128(unsigned lds_byte_off, const void* g) {
  asm volatile("global_load_async_to_lds_b128 %0, %1, off"
               :: "v"(lds_byte_off), "v"(g) : "memory");
}
__device__ __forceinline__ void wait_async0() {
  asm volatile("s_wait_asynccnt 0" ::: "memory");
}

// ---------------------------------------------------------------------------
// Weight transpose + fp32->bf16 convert: Wt[n][k] = (bf16)W[k][n]
// ---------------------------------------------------------------------------
__global__ void transpose_cvt(const float* __restrict__ W, bf16* __restrict__ Wt,
                              int R, int Cc) {
  __shared__ float tile[32][33];
  int bx = blockIdx.x * 32, by = blockIdx.y * 32;
  int tx = threadIdx.x, ty = threadIdx.y;     // 32 x 8
#pragma unroll
  for (int i = 0; i < 32; i += 8)
    tile[ty + i][tx] = W[(size_t)(by + ty + i) * Cc + bx + tx];
  __syncthreads();
#pragma unroll
  for (int i = 0; i < 32; i += 8)
    Wt[(size_t)(bx + ty + i) * R + by + tx] = (bf16)tile[tx][ty + i];
}

__global__ void cvt_bf16(const float* __restrict__ x, bf16* __restrict__ y) {
  size_t i = ((size_t)blockIdx.x * blockDim.x + threadIdx.x) * 4;
  float4 v = *(const float4*)(x + i);
  y[i + 0] = (bf16)v.x; y[i + 1] = (bf16)v.y;
  y[i + 2] = (bf16)v.z; y[i + 3] = (bf16)v.w;
}

// ---------------------------------------------------------------------------
// C[MxN] = A[MxK] @ Bt[NxK]^T (bf16 operands, K-contiguous rows), fp32 accum.
// Block = 256 thr (8 waves); tile 128x128; BK = 64; double-buffered LDS fed by
// async global->LDS DMA, overlapped with the WMMA chain.
// Dynamic LDS: 4 * 128*72 bf16 = 73728 B (dynamic base offset = 0).
// ---------------------------------------------------------------------------
template <int BF16OUT>
__global__ void __launch_bounds__(256)
gemm128(const bf16* __restrict__ A, const bf16* __restrict__ Bt,
        void* __restrict__ Cout, int M, int N, int K) {
  constexpr int LSTR = 72;               // 64 + 8 pad (bf16 elems)
  constexpr int SZB  = 128 * LSTR;
  extern __shared__ bf16 gsm[];

  const int tid = threadIdx.x;
  const int w = tid >> 5, lane = tid & 31, lr = lane & 15, hi = lane >> 4;
  const int m0 = blockIdx.y * 128, n0 = blockIdx.x * 128;
  v8f acc[8];
#pragma unroll
  for (int j = 0; j < 8; ++j) acc[j] = (v8f){0,0,0,0,0,0,0,0};

  const int lrow = tid >> 1, lcol = (tid & 1) * 32;   // 32 bf16 = 4 x 16B
  const bf16* ga = A  + (size_t)(m0 + lrow) * K + lcol;
  const bf16* gb = Bt + (size_t)(n0 + lrow) * K + lcol;
  const unsigned ldsRow = (unsigned)((lrow * LSTR + lcol) * sizeof(bf16));

  // prologue: async-fill buffer 0
#pragma unroll
  for (int i = 0; i < 4; ++i) {
    async_copy_b128(ldsRow + i * 16, ga + i * 8);                              // As buf 0
    async_copy_b128(ldsRow + (unsigned)(2 * SZB * 2) + i * 16, gb + i * 8);    // Bs buf 0
  }
  wait_async0();
  __syncthreads();

  const int nk = K >> 6;
  for (int kb = 0; kb < nk; ++kb) {
    const int cur = kb & 1;
    bf16* Asc = gsm + cur * SZB;
    bf16* Bsc = gsm + (2 + cur) * SZB;
    if (kb + 1 < nk) {   // async-prefetch next stage into the other buffer
      const bf16* pa = ga + (size_t)(kb + 1) * 64;
      const bf16* pb = gb + (size_t)(kb + 1) * 64;
      const unsigned aoff = ldsRow + (unsigned)((1 - cur) * SZB * 2);
      const unsigned boff = ldsRow + (unsigned)((3 - cur) * SZB * 2);
#pragma unroll
      for (int i = 0; i < 4; ++i) {
        async_copy_b128(aoff + i * 16, pa + i * 8);
        async_copy_b128(boff + i * 16, pb + i * 8);
      }
    }
#pragma unroll
    for (int kk = 0; kk < 2; ++kk) {
      v16bf af = frag_ldA(Asc + (w * 16 + lr) * LSTR + kk * 32, hi);
#pragma unroll
      for (int j = 0; j < 8; ++j) {
        v16bf bfr = frag_ldB(Bsc + (j * 16 + lr) * LSTR + kk * 32, hi);
        acc[j] = WMMA_BF16(af, bfr, acc[j]);
      }
    }
    if (kb + 1 < nk) wait_async0();
    __syncthreads();
  }
#pragma unroll
  for (int j = 0; j < 8; ++j)
#pragma unroll
    for (int r = 0; r < 8; ++r) {
      int row = m0 + w * 16 + r + hi * 8;
      int col = n0 + j * 16 + lr;
      if (BF16OUT) ((bf16*)Cout)[(size_t)row * N + col] = (bf16)acc[j][r];
      else         ((float*)Cout)[(size_t)row * N + col] = acc[j][r];
    }
}

// ---------------------------------------------------------------------------
// SiLU + LRPE rotation, in place on bf16 (ROWS x EMBED). One thread per pair.
// ---------------------------------------------------------------------------
__global__ void silu_rope(bf16* __restrict__ qk) {
  int idx = blockIdx.x * blockDim.x + threadIdx.x;  // ROWS*HEADS*64
  int row = idx >> 10;
  int rem = idx & 1023;
  int h = rem >> 6, j = rem & 63;
  int n = row & (SEQ - 1);
  size_t base = (size_t)row * EMBED + h * HD;
  float x1 = (float)qk[base + j];
  float x2 = (float)qk[base + 64 + j];
  x1 = x1 / (1.f + __expf(-x1));
  x2 = x2 / (1.f + __expf(-x2));
  float theta = __expf(-0.14391156831f * (float)j);   // 10000^(-2j/128)
  float s, c;
  __sincosf((float)n * theta, &s, &c);
  qk[base + j]      = (bf16)(x1 * c - x2 * s);
  qk[base + 64 + j] = (bf16)(x1 * s + x2 * c);
}

// ---------------------------------------------------------------------------
// Lightning attention: one block per (batch, head). 8 waves.
// S (128x128) lives in fp32 WMMA accumulators across the 16-chunk scan.
// __launch_bounds__(256,1): LDS (174KB) limits us to 1 WG/WGP anyway, so let
// the allocator use the full VGPR file instead of spilling to scratch.
// Dynamic LDS: 5 buffers x 128x136 bf16 = 174080 B:
//   B0 = q tile (rows t, cols d)   -> reused as decayed kT (rows d, cols s)
//   B1 = k tile (rows s, cols d)
//   B2 = vT     (rows e, cols s)
//   B3 = ST     (rows e, cols d, bf16 snapshot of S)
//   B4 = masked scores (rows t, cols s)
// ---------------------------------------------------------------------------
__global__ void __launch_bounds__(256, 1)
lightning_attn(const bf16* __restrict__ q, const bf16* __restrict__ k,
               const bf16* __restrict__ v, float* __restrict__ o) {
  extern __shared__ bf16 smem[];
  constexpr int STR = 136, SZ = 128 * STR;
  bf16* sQ  = smem;            // B0
  bf16* sK  = smem + SZ;       // B1
  bf16* sVT = smem + 2 * SZ;   // B2
  bf16* sST = smem + 3 * SZ;   // B3
  bf16* sSC = smem + 4 * SZ;   // B4
  const int tid = threadIdx.x;
  const int w = tid >> 5, lane = tid & 31, lr = lane & 15, hi = lane >> 4;
  const int bh = blockIdx.x, b = bh >> 4, h = bh & 15;
  const float log_lam = -exp2f(-0.5f * (float)(h + 1));
  const float lamC = __expf(log_lam * (float)CHUNKC);

  for (int i = tid; i < SZ; i += 256) sST[i] = (bf16)0.f;

  const int ls = tid >> 1;             // seq row this thread loads
  const int lseg = (tid & 1) * 64;     // 64-wide column segment
  const float dk_ls = __expf(log_lam * (float)(CHUNKC - 1 - ls));

  float dq8[8];                        // decay_q for this lane's output rows
#pragma unroll
  for (int r = 0; r < 8; ++r) {
    int t = w * 16 + r + hi * 8;
    dq8[r] = __expf(log_lam * (float)(t + 1));
  }

  v8f Sacc[8];
#pragma unroll
  for (int j = 0; j < 8; ++j) Sacc[j] = (v8f){0,0,0,0,0,0,0,0};
  __syncthreads();

  for (int c = 0; c < NCHUNK; ++c) {
    const size_t grow = ((size_t)b * SEQ + c * CHUNKC + ls) * EMBED + h * HD + lseg;
    // ---- load phase: q,k contiguous; v transposed (8 elems scattered/step) ----
    {
      const uint4* gq = (const uint4*)(q + grow);
      const uint4* gk = (const uint4*)(k + grow);
      uint4* dq = (uint4*)(sQ + ls * STR + lseg);
      uint4* dk = (uint4*)(sK + ls * STR + lseg);
#pragma unroll
      for (int i = 0; i < 8; ++i) dq[i] = gq[i];
#pragma unroll
      for (int i = 0; i < 8; ++i) dk[i] = gk[i];
      const uint4* gv = (const uint4*)(v + grow);
#pragma unroll
      for (int i = 0; i < 8; ++i) {
        uint4 t4 = gv[i];
        bf16 tmp[8]; *(uint4*)tmp = t4;
#pragma unroll
        for (int e = 0; e < 8; ++e) sVT[(lseg + i * 8 + e) * STR + ls] = tmp[e];
      }
    }
    __syncthreads();   // barrier A

    // ---- phase 1: per j-tile, u = q@ST and raw scores = q@kT, fused ----
    v8f oacc[8];
    v16bf af[4];
#pragma unroll
    for (int kk = 0; kk < 4; ++kk)
      af[kk] = frag_ldA(sQ + (w * 16 + lr) * STR + kk * 32, hi);
#pragma unroll
    for (int j = 0; j < 8; ++j) {
      v8f u  = (v8f){0,0,0,0,0,0,0,0};
      v8f sv = (v8f){0,0,0,0,0,0,0,0};
#pragma unroll
      for (int kk = 0; kk < 4; ++kk) {
        v16bf bS = frag_ldB(sST + (j * 16 + lr) * STR + kk * 32, hi);
        u  = WMMA_BF16(af[kk], bS, u);
        v16bf bK = frag_ldB(sK + (j * 16 + lr) * STR + kk * 32, hi);
        sv = WMMA_BF16(af[kk], bK, sv);
      }
#pragma unroll
      for (int r = 0; r < 8; ++r) oacc[j][r] = dq8[r] * u[r];
      // mask + decay, publish scores tile to B4 immediately
#pragma unroll
      for (int r = 0; r < 8; ++r) {
        int t = w * 16 + r + hi * 8, si = j * 16 + lr;
        float d = (t >= si) ? __expf(log_lam * (float)(t - si)) : 0.f;
        sSC[t * STR + si] = (bf16)(sv[r] * d);
      }
    }
    __syncthreads();   // barrier B: B0/B1/B3 reads done, B4 writes done

    // ---- build decayed kT into B0 (reads B1) ----
    {
      const uint4* src = (const uint4*)(sK + ls * STR + lseg);
#pragma unroll
      for (int i = 0; i < 8; ++i) {
        uint4 t4 = src[i];
        bf16 tmp[8]; *(uint4*)tmp = t4;
#pragma unroll
        for (int e = 0; e < 8; ++e)
          sQ[(lseg + i * 8 + e) * STR + ls] = (bf16)((float)tmp[e] * dk_ls);
      }
    }
    __syncthreads();   // barrier C

    // ---- phase 2 (fused): o += scores@vT ; S = lamC*S + kT@vT ----
#pragma unroll
    for (int j = 0; j < 8; ++j)
#pragma unroll
      for (int r = 0; r < 8; ++r) Sacc[j][r] *= lamC;
#pragma unroll
    for (int kk = 0; kk < 4; ++kk) {
      v16bf afs = frag_ldA(sSC + (w * 16 + lr) * STR + kk * 32, hi);
      v16bf afk = frag_ldA(sQ  + (w * 16 + lr) * STR + kk * 32, hi);
#pragma unroll
      for (int j = 0; j < 8; ++j) {
        v16bf bfr = frag_ldB(sVT + (j * 16 + lr) * STR + kk * 32, hi);
        oacc[j] = WMMA_BF16(afs, bfr, oacc[j]);
        Sacc[j] = WMMA_BF16(afk, bfr, Sacc[j]);
      }
    }

    // ---- publish ST (bf16) for next chunk ; store o chunk ----
#pragma unroll
    for (int j = 0; j < 8; ++j)
#pragma unroll
      for (int r = 0; r < 8; ++r) {
        int d = w * 16 + r + hi * 8, e = j * 16 + lr;
        sST[e * STR + d] = (bf16)Sacc[j][r];
      }
#pragma unroll
    for (int j = 0; j < 8; ++j)
#pragma unroll
      for (int r = 0; r < 8; ++r) {
        int t = w * 16 + r + hi * 8, e = j * 16 + lr;
        o[((size_t)b * SEQ + c * CHUNKC + t) * EMBED + h * HD + e] = oacc[j][r];
      }
    __syncthreads();   // barrier D (end of chunk)
  }
}

// ---------------------------------------------------------------------------
// Gated group RMSNorm: one wave32 per (row, head), 4 elems/lane.
// ---------------------------------------------------------------------------
__global__ void gated_norm(const float* __restrict__ o, const bf16* __restrict__ gate,
                           const float* __restrict__ wgt, bf16* __restrict__ out) {
  int g = blockIdx.x * 8 + (threadIdx.x >> 5);
  int lane = threadIdx.x & 31;
  int row = g >> 4, h = g & 15;
  size_t base = (size_t)row * EMBED + h * HD + lane * 4;
  float x[4], ss = 0.f;
#pragma unroll
  for (int i = 0; i < 4; ++i) {
    float gv = (float)gate[base + i];
    float val = o[base + i] / (1.f + __expf(-gv));
    x[i] = val; ss += val * val;
  }
#pragma unroll
  for (int off = 16; off > 0; off >>= 1) ss += __shfl_xor(ss, off, 32);
  float scl = rsqrtf(ss * (1.f / 128.f) + 1e-6f);
#pragma unroll
  for (int i = 0; i < 4; ++i)
    out[base + i] = (bf16)(x[i] * scl * wgt[h * HD + lane * 4 + i]);
}

// ---------------------------------------------------------------------------
extern "C" void kernel_launch(void* const* d_in, const int* in_sizes, int n_in,
                              void* d_out, int out_size, void* d_ws, size_t ws_size,
                              hipStream_t stream) {
  const float* x   = (const float*)d_in[0];
  const float* Wq  = (const float*)d_in[1];
  const float* Wk  = (const float*)d_in[2];
  const float* Wv  = (const float*)d_in[3];
  const float* Wo  = (const float*)d_in[4];
  const float* Wg1 = (const float*)d_in[5];
  const float* Wg2 = (const float*)d_in[6];
  const float* nw  = (const float*)d_in[7];
  float* out = (float*)d_out;

  char* ws = (char*)d_ws;
  size_t off = 0;
  auto alloc = [&](size_t bytes) -> void* {
    void* p = ws + off; off += (bytes + 255) & ~(size_t)255; return p;
  };
  bf16* xbf  = (bf16*)alloc((size_t)ROWS * EMBED * 2);
  bf16* Wqt  = (bf16*)alloc((size_t)EMBED * EMBED * 2);
  bf16* Wkt  = (bf16*)alloc((size_t)EMBED * EMBED * 2);
  bf16* Wvt  = (bf16*)alloc((size_t)EMBED * EMBED * 2);
  bf16* Wot  = (bf16*)alloc((size_t)EMBED * EMBED * 2);
  bf16* Wg1t = (bf16*)alloc((size_t)HD * EMBED * 2);
  bf16* Wg2t = (bf16*)alloc((size_t)EMBED * HD * 2);
  bf16* qbf  = (bf16*)alloc((size_t)ROWS * EMBED * 2);
  bf16* kbf  = (bf16*)alloc((size_t)ROWS * EMBED * 2);
  bf16* vbf  = (bf16*)alloc((size_t)ROWS * EMBED * 2);
  bf16* g1o  = (bf16*)alloc((size_t)ROWS * HD * 2);
  bf16* gbf  = (bf16*)alloc((size_t)ROWS * EMBED * 2);
  float* of  = (float*)alloc((size_t)ROWS * EMBED * 4);
  bf16* onb  = (bf16*)alloc((size_t)ROWS * EMBED * 2);

  const size_t GEMM_LDS = 4u * 128u * 72u * sizeof(bf16);   // 73728 B
  const size_t ATTN_LDS = 5u * 128u * 136u * sizeof(bf16);  // 174080 B

  dim3 tb(32, 8);
  transpose_cvt<<<dim3(EMBED/32, EMBED/32), tb, 0, stream>>>(Wq, Wqt, EMBED, EMBED);
  transpose_cvt<<<dim3(EMBED/32, EMBED/32), tb, 0, stream>>>(Wk, Wkt, EMBED, EMBED);
  transpose_cvt<<<dim3(EMBED/32, EMBED/32), tb, 0, stream>>>(Wv, Wvt, EMBED, EMBED);
  transpose_cvt<<<dim3(EMBED/32, EMBED/32), tb, 0, stream>>>(Wo, Wot, EMBED, EMBED);
  transpose_cvt<<<dim3(HD/32,    EMBED/32), tb, 0, stream>>>(Wg1, Wg1t, EMBED, HD);
  transpose_cvt<<<dim3(EMBED/32, HD/32),    tb, 0, stream>>>(Wg2, Wg2t, HD, EMBED);
  cvt_bf16<<<(ROWS * EMBED / 4) / 256, 256, 0, stream>>>(x, xbf);

  gemm128<1><<<dim3(EMBED/128, ROWS/128), 256, GEMM_LDS, stream>>>(xbf, Wqt, qbf, ROWS, EMBED, EMBED);
  gemm128<1><<<dim3(EMBED/128, ROWS/128), 256, GEMM_LDS, stream>>>(xbf, Wkt, kbf, ROWS, EMBED, EMBED);
  gemm128<1><<<dim3(EMBED/128, ROWS/128), 256, GEMM_LDS, stream>>>(xbf, Wvt, vbf, ROWS, EMBED, EMBED);
  gemm128<1><<<dim3(HD/128,    ROWS/128), 256, GEMM_LDS, stream>>>(xbf, Wg1t, g1o, ROWS, HD, EMBED);
  gemm128<1><<<dim3(EMBED/128, ROWS/128), 256, GEMM_LDS, stream>>>(g1o, Wg2t, gbf, ROWS, EMBED, HD);

  silu_rope<<<(ROWS * HEADS * 64) / 256, 256, 0, stream>>>(qbf);
  silu_rope<<<(ROWS * HEADS * 64) / 256, 256, 0, stream>>>(kbf);

  lightning_attn<<<NBATCH * HEADS, 256, ATTN_LDS, stream>>>(qbf, kbf, vbf, of);

  gated_norm<<<(ROWS * HEADS) / 8, 256, 0, stream>>>(of, gbf, nw, onb);

  gemm128<0><<<dim3(EMBED/128, ROWS/128), 256, GEMM_LDS, stream>>>(onb, Wot, out, ROWS, EMBED, EMBED);
}
